// OTLayer_47253230190781
// MI455X (gfx1250) — matrix operand order repaired and checked by reading, not compiled, not run on hardware
//
#include <hip/hip_runtime.h>

// ---------------------------------------------------------------------------
// Types for CDNA5 WMMA (wave32)
// ---------------------------------------------------------------------------
typedef __attribute__((ext_vector_type(16))) __bf16 v16bf;
typedef __attribute__((ext_vector_type(8)))  __bf16 v8bf;
typedef __attribute__((ext_vector_type(8)))  float  v8f;

// GCC-style int4 vector matching the async-LDS builtin's pointee type
typedef int v4i_g __attribute__((vector_size(16)));
typedef __attribute__((address_space(1))) v4i_g* gv4p;   // global (AS1)
typedef __attribute__((address_space(3))) v4i_g* lv4p;   // LDS (AS3)

// ---------------------------------------------------------------------------
// Problem constants
// ---------------------------------------------------------------------------
#define NB      16      // batch n
#define INS     1024    // in_size
#define IND     1024    // in_dim
#define HEADS   4
#define OUTS    128     // out_size
#define OUTD    1024    // out_dim
#define CDIM    4096    // heads * in_dim
#define INV_EPS 10.0f
#define LOG_A   (-2.0794415416798357f)   // log(out_size/in_size) = log(1/8)

// LDS tile: 128 rows x 32 K-elems, rows padded to 40 elems (80B) for bank spread
#define LDS_LD  40
#define TILE_E  (128 * LDS_LD)

#if __has_builtin(__builtin_amdgcn_global_load_async_to_lds_b128) && \
    __has_builtin(__builtin_amdgcn_s_wait_asynccnt)
#define HAVE_ASYNC_LDS 1
#else
#define HAVE_ASYNC_LDS 0
#endif

// ---------------------------------------------------------------------------
// WMMA helpers
// ---------------------------------------------------------------------------
__device__ __forceinline__ v8f wmma_bf16(v16bf a, v16bf b, v8f c) {
  return __builtin_amdgcn_wmma_f32_16x16x32_bf16(
      /*neg_a=*/false, a, /*neg_b=*/false, b,
      /*c_mod=*/(short)0, c, /*reuse_a=*/false, /*reuse_b=*/false);
}

// A fragment, 16x32 bf16 (M x K). ISA layout:
// lanes 0-15 : M=lane,    K = 0..7 and 16..23
// lanes 16-31: M=lane-16, K = 8..15 and 24..31
__device__ __forceinline__ v16bf load_fragA(const __bf16* base, int ld, int lane) {
  const int half = lane >> 4;
  const int r    = lane & 15;
  const __bf16* p = base + (size_t)r * ld + half * 8;
  v8bf lo = *(const v8bf*)(p);        // K = half*8 .. +7
  v8bf hi = *(const v8bf*)(p + 16);   // K = 16 + half*8 .. +7
  v16bf out;
#pragma unroll
  for (int j = 0; j < 8; ++j) { out[j] = lo[j]; out[j + 8] = hi[j]; }
  return out;
}

// B fragment, 32x16 bf16 (K x N), sourced from row-major B^T (N rows, K contig).
// Layout: lanes 0-15 hold K=0..7,16..23 of column lane; lanes 16-31 hold K=8..15,24..31.
__device__ __forceinline__ v16bf load_fragB(const __bf16* baseT, int ld, int lane) {
  const int half = lane >> 4;
  const int c    = lane & 15;
  const __bf16* p = baseT + (size_t)c * ld + half * 8;
  v8bf lo = *(const v8bf*)(p);
  v8bf hi = *(const v8bf*)(p + 16);
  v16bf out;
#pragma unroll
  for (int j = 0; j < 8; ++j) { out[j] = lo[j]; out[j + 8] = hi[j]; }
  return out;
}

// ---------------------------------------------------------------------------
// Tile staging: 128 x 32 bf16 (64B/row) global -> LDS. 512 b128 ops, 2/thread.
// Async path uses GLOBAL_LOAD_ASYNC_TO_LDS_B128 (ASYNCcnt), else load+ds_store.
// ---------------------------------------------------------------------------
__device__ __forceinline__ void stage_tile(const __bf16* __restrict__ g, int ld,
                                           __bf16* lds, int t) {
#pragma unroll
  for (int o = 0; o < 2; ++o) {
    const int idx = t + o * 256;
    const int r = idx >> 2;
    const int c = idx & 3;
    const __bf16* src = g + (size_t)r * ld + c * 8;
    __bf16* dst = lds + r * LDS_LD + c * 8;
#if HAVE_ASYNC_LDS
    __builtin_amdgcn_global_load_async_to_lds_b128((gv4p)src, (lv4p)dst, 0, 0);
#else
    *(v8bf*)dst = *(const v8bf*)src;
#endif
  }
}

__device__ __forceinline__ void wait_stage() {
#if HAVE_ASYNC_LDS
  __builtin_amdgcn_s_wait_asynccnt(0);
#endif
  __syncthreads();
}

// ---------------------------------------------------------------------------
// Shared GEMM core: block tile 128(M) x 128(N), k-step 32, double-buffered LDS.
// 8 waves: waveM in 0..3 (32 rows each), waveN in 0..1 (64 cols each).
// ---------------------------------------------------------------------------
__device__ __forceinline__ void gemm_core(
    const __bf16* __restrict__ Ag, int lda,
    const __bf16* __restrict__ Bg, int ldb, int kdim,
    __bf16* ldsA, __bf16* ldsB, v8f acc[8],
    int t, int lane, int waveM, int waveN) {
  stage_tile(Ag, lda, ldsA, t);
  stage_tile(Bg, ldb, ldsB, t);
  for (int k = 0; k < kdim; k += 32) {
    const int cur = (k >> 5) & 1;
    wait_stage();                       // tile k resident in buf[cur]
    if (k + 32 < kdim) {                // overlap copy of tile k+1 with compute
      stage_tile(Ag + k + 32, lda, ldsA + (cur ^ 1) * TILE_E, t);
      stage_tile(Bg + k + 32, ldb, ldsB + (cur ^ 1) * TILE_E, t);
    }
    const __bf16* As = ldsA + cur * TILE_E + (waveM * 32) * LDS_LD;
    const __bf16* Bs = ldsB + cur * TILE_E + (waveN * 64) * LDS_LD;
    v16bf a0 = load_fragA(As,               LDS_LD, lane);
    v16bf a1 = load_fragA(As + 16 * LDS_LD, LDS_LD, lane);
    v16bf b0 = load_fragB(Bs,               LDS_LD, lane);
    v16bf b1 = load_fragB(Bs + 16 * LDS_LD, LDS_LD, lane);
    v16bf b2 = load_fragB(Bs + 32 * LDS_LD, LDS_LD, lane);
    v16bf b3 = load_fragB(Bs + 48 * LDS_LD, LDS_LD, lane);
    acc[0] = wmma_bf16(a0, b0, acc[0]);
    acc[1] = wmma_bf16(a0, b1, acc[1]);
    acc[2] = wmma_bf16(a0, b2, acc[2]);
    acc[3] = wmma_bf16(a0, b3, acc[3]);
    acc[4] = wmma_bf16(a1, b0, acc[4]);
    acc[5] = wmma_bf16(a1, b1, acc[5]);
    acc[6] = wmma_bf16(a1, b2, acc[6]);
    acc[7] = wmma_bf16(a1, b3, acc[7]);
    __syncthreads();                    // all reads of buf[cur] done before reuse
  }
}

// ---------------------------------------------------------------------------
// Conversion kernels (f32 -> bf16), x additionally transposed through LDS
// ---------------------------------------------------------------------------
__global__ __launch_bounds__(256) void convert_flat_kernel(
    const float* __restrict__ in, __bf16* __restrict__ out, int nelem) {
  int idx = (blockIdx.x * blockDim.x + threadIdx.x) * 4;
  if (idx + 3 < nelem) {
    float4 f = *(const float4*)(in + idx);
    out[idx + 0] = (__bf16)f.x;
    out[idx + 1] = (__bf16)f.y;
    out[idx + 2] = (__bf16)f.z;
    out[idx + 3] = (__bf16)f.w;
  }
}

// grid (IND/32, INS/32, NB), block (32, 8)
__global__ __launch_bounds__(256) void convert_x_kernel(
    const float* __restrict__ x, __bf16* __restrict__ xb, __bf16* __restrict__ xTb) {
  __shared__ __bf16 tile[32][33];
  const int n  = blockIdx.z;
  const int d0 = blockIdx.x * 32;
  const int i0 = blockIdx.y * 32;
  const int tx = threadIdx.x, ty = threadIdx.y;
  const float* src = x + ((size_t)n * INS + i0) * IND + d0;
  __bf16* dst = xb + ((size_t)n * INS + i0) * IND + d0;
#pragma unroll
  for (int rr = 0; rr < 32; rr += 8) {
    __bf16 h = (__bf16)src[(size_t)(ty + rr) * IND + tx];
    dst[(size_t)(ty + rr) * IND + tx] = h;
    tile[ty + rr][tx] = h;
  }
  __syncthreads();
  __bf16* dstT = xTb + ((size_t)n * IND + d0) * INS + i0;
#pragma unroll
  for (int rr = 0; rr < 32; rr += 8) {
    dstT[(size_t)(ty + rr) * INS + tx] = tile[tx][ty + rr];
  }
}

// ---------------------------------------------------------------------------
// GEMM1: Kmat[n,m,i,s] = (x[n,i,:] . otw[m,s,:]) / eps
// C[16384 x 512] = A(x_bf16) * B^T(otw_bf16), K=1024.
// ---------------------------------------------------------------------------
__global__ __launch_bounds__(256) void gemm1_kernel(
    const __bf16* __restrict__ A, const __bf16* __restrict__ Bt,
    float* __restrict__ Kout) {
  __shared__ __bf16 ldsA[2 * TILE_E];
  __shared__ __bf16 ldsB[2 * TILE_E];
  const int t     = threadIdx.x;
  const int lane  = t & 31;
  const int wave  = t >> 5;
  const int waveM = wave >> 1;
  const int waveN = wave & 1;
  const int row0  = blockIdx.x * 128;
  const int col0  = blockIdx.y * 128;

  const v8f z = {0.f, 0.f, 0.f, 0.f, 0.f, 0.f, 0.f, 0.f};
  v8f acc[8] = {z, z, z, z, z, z, z, z};

  gemm_core(A + (size_t)row0 * IND, IND, Bt + (size_t)col0 * IND, IND, IND,
            ldsA, ldsB, acc, t, lane, waveM, waveN);

  const int colLane = lane & 15;
  const int rowHalf = (lane >> 4) * 8;
#pragma unroll
  for (int mi = 0; mi < 2; ++mi)
#pragma unroll
    for (int ni = 0; ni < 4; ++ni) {
      v8f c = acc[mi * 4 + ni];
#pragma unroll
      for (int r = 0; r < 8; ++r) {
        int row = row0 + waveM * 32 + mi * 16 + r + rowHalf;  // = n*1024 + i
        int col = col0 + waveN * 64 + ni * 16 + colLane;      // = m*128 + s
        int n = row >> 10, i = row & 1023;
        int m = col >> 7,  s = col & 127;
        size_t idx = (((size_t)n * HEADS + m) * INS + i) * OUTS + s;
        Kout[idx] = c[r] * INV_EPS;
      }
    }
}

// ---------------------------------------------------------------------------
// Sinkhorn (log domain, 10 iterations) per batch b = n*4+m.
// One workgroup (256 thr) per batch; u[1024], v[128] in LDS.
// Emits T^T in bf16: Tt[b][s][i] = exp(K[b][i][s] + u[i] + v[s]).
// ---------------------------------------------------------------------------
__global__ __launch_bounds__(256) void sinkhorn_kernel(
    const float* __restrict__ Kmat, __bf16* __restrict__ Tt) {
  const int b = blockIdx.x;
  const float* Kb = Kmat + (size_t)b * INS * OUTS;
  __shared__ float u[INS];
  __shared__ float v[OUTS];
  __shared__ float redm[256];
  __shared__ float reds[256];
  const int t    = threadIdx.x;
  const int lane = t & 31;
  const int wave = t >> 5;

  for (int i = t; i < INS; i += 256) u[i] = 0.f;
  if (t < OUTS) v[t] = 0.f;
  __syncthreads();

  for (int it = 0; it < 10; ++it) {
    // ---- u update: u[i] = a - (u[i] + lse_s(K[i,s] + v[s]))
    for (int i = wave; i < INS; i += 8) {
      const float* row = Kb + (size_t)i * OUTS;
      float x0 = row[lane]      + v[lane];
      float x1 = row[lane + 32] + v[lane + 32];
      float x2 = row[lane + 64] + v[lane + 64];
      float x3 = row[lane + 96] + v[lane + 96];
      float m = fmaxf(fmaxf(x0, x1), fmaxf(x2, x3));
#pragma unroll
      for (int off = 16; off; off >>= 1) m = fmaxf(m, __shfl_xor(m, off, 32));
      float s = __expf(x0 - m) + __expf(x1 - m) + __expf(x2 - m) + __expf(x3 - m);
#pragma unroll
      for (int off = 16; off; off >>= 1) s += __shfl_xor(s, off, 32);
      if (lane == 0) u[i] = LOG_A - (u[i] + m + __logf(s));
    }
    __syncthreads();

    // ---- v update: v[s] = -(v[s] + lse_i(K[i,s] + u[i])), online LSE
    {
      const int sc   = t & 127;
      const int half = t >> 7;
      float m = -3.0e38f, ssum = 0.f;
      const float* col = Kb + sc + (size_t)half * 512 * OUTS;
      const float* uu  = u + half * 512;
      for (int i = 0; i < 512; ++i) {
        float x = col[(size_t)i * OUTS] + uu[i];
        if (x > m) { ssum = ssum * __expf(m - x) + 1.f; m = x; }
        else       { ssum += __expf(x - m); }
      }
      redm[t] = m; reds[t] = ssum;
    }
    __syncthreads();
    if (t < OUTS) {
      float m0 = redm[t], s0 = reds[t];
      float m1 = redm[t + 128], s1 = reds[t + 128];
      float m = fmaxf(m0, m1);
      float s = s0 * __expf(m0 - m) + s1 * __expf(m1 - m);
      v[t] = -(v[t] + m + __logf(s));
    }
    __syncthreads();
  }

  // ---- emit transposed transport plan in bf16 (coalesced stores)
  __bf16* Tb = Tt + (size_t)b * INS * OUTS;
  for (int idx = t; idx < INS * OUTS; idx += 256) {
    int i = idx & 1023;        // fast dim of Tt row
    int s = idx >> 10;
    float val = __expf(Kb[(size_t)i * OUTS + s] + u[i] + v[s]);
    Tb[idx] = (__bf16)val;     // Tt[b][s][i]
  }
}

// ---------------------------------------------------------------------------
// GEMM2 (batched over bm = n*4+m):
//   feat[n][s][m][d] = sum_i Tt[bm][s][i] * xT[n][d][i]
// A = Tt[bm] (128 x 1024), B^T = xT[n] (1024 x 1024), K = 1024. Output bf16.
// grid (1, IND/128, 64), block 256.
// ---------------------------------------------------------------------------
__global__ __launch_bounds__(256) void gemm2_kernel(
    const __bf16* __restrict__ Tt, const __bf16* __restrict__ xT,
    __bf16* __restrict__ feat) {
  __shared__ __bf16 ldsA[2 * TILE_E];
  __shared__ __bf16 ldsB[2 * TILE_E];
  const int t     = threadIdx.x;
  const int lane  = t & 31;
  const int wave  = t >> 5;
  const int waveM = wave >> 1;
  const int waveN = wave & 1;
  const int bm = blockIdx.z;
  const int n  = bm >> 2;
  const int m  = bm & 3;
  const int col0 = blockIdx.y * 128;

  const v8f z = {0.f, 0.f, 0.f, 0.f, 0.f, 0.f, 0.f, 0.f};
  v8f acc[8] = {z, z, z, z, z, z, z, z};

  gemm_core(Tt + (size_t)bm * OUTS * INS, INS,
            xT + ((size_t)n * IND + col0) * INS, INS, INS,
            ldsA, ldsB, acc, t, lane, waveM, waveN);

  const int colLane = lane & 15;
  const int rowHalf = (lane >> 4) * 8;
#pragma unroll
  for (int mi = 0; mi < 2; ++mi)
#pragma unroll
    for (int ni = 0; ni < 4; ++ni) {
      v8f c = acc[mi * 4 + ni];
#pragma unroll
      for (int r = 0; r < 8; ++r) {
        int s = waveM * 32 + mi * 16 + r + rowHalf;
        int d = col0 + waveN * 64 + ni * 16 + colLane;
        size_t idx = (((size_t)n * OUTS + s) * HEADS + m) * IND + d;
        feat[idx] = (__bf16)c[r];
      }
    }
}

// ---------------------------------------------------------------------------
// GEMM3: out[row][o] = relu(sum_c feat[row][c] * linw[o][c] + linb[o])
// A = feat (2048 x 4096), B^T = linw (1024 x 4096), K = 4096. Output f32.
// grid (2048/128, 1024/128), block 256.
// ---------------------------------------------------------------------------
__global__ __launch_bounds__(256) void gemm3_kernel(
    const __bf16* __restrict__ A, const __bf16* __restrict__ Bt,
    const float* __restrict__ bias, float* __restrict__ out) {
  __shared__ __bf16 ldsA[2 * TILE_E];
  __shared__ __bf16 ldsB[2 * TILE_E];
  const int t     = threadIdx.x;
  const int lane  = t & 31;
  const int wave  = t >> 5;
  const int waveM = wave >> 1;
  const int waveN = wave & 1;
  const int row0  = blockIdx.x * 128;
  const int col0  = blockIdx.y * 128;

  const v8f z = {0.f, 0.f, 0.f, 0.f, 0.f, 0.f, 0.f, 0.f};
  v8f acc[8] = {z, z, z, z, z, z, z, z};

  gemm_core(A + (size_t)row0 * CDIM, CDIM, Bt + (size_t)col0 * CDIM, CDIM, CDIM,
            ldsA, ldsB, acc, t, lane, waveM, waveN);

  const int colLane = lane & 15;
  const int rowHalf = (lane >> 4) * 8;
#pragma unroll
  for (int mi = 0; mi < 2; ++mi)
#pragma unroll
    for (int ni = 0; ni < 4; ++ni) {
      v8f c = acc[mi * 4 + ni];
#pragma unroll
      for (int r = 0; r < 8; ++r) {
        int row = row0 + waveM * 32 + mi * 16 + r + rowHalf;
        int col = col0 + waveN * 64 + ni * 16 + colLane;
        float val = c[r] + bias[col];
        out[(size_t)row * OUTD + col] = fmaxf(val, 0.f);
      }
    }
}

// ---------------------------------------------------------------------------
// Launch pipeline
// ---------------------------------------------------------------------------
extern "C" void kernel_launch(void* const* d_in, const int* in_sizes, int n_in,
                              void* d_out, int out_size, void* d_ws, size_t ws_size,
                              hipStream_t stream) {
  const float* x    = (const float*)d_in[0];   // [16,1024,1024]
  const float* otw  = (const float*)d_in[1];   // [4,128,1024]
  const float* linw = (const float*)d_in[2];   // [1024,4096]
  const float* linb = (const float*)d_in[3];   // [1024]

  char* ws = (char*)d_ws;
  size_t off = 0;
  __bf16* x_bf    = (__bf16*)(ws + off); off += (size_t)NB * INS * IND * 2;      // 33.5 MB
  __bf16* xT_bf   = (__bf16*)(ws + off); off += (size_t)NB * IND * INS * 2;      // 33.5 MB
  __bf16* otw_bf  = (__bf16*)(ws + off); off += (size_t)HEADS * OUTS * IND * 2;  // 1 MB
  __bf16* linw_bf = (__bf16*)(ws + off); off += (size_t)OUTD * CDIM * 2;         // 8 MB
  float*  Kmat    = (float*) (ws + off); off += (size_t)NB * HEADS * INS * OUTS * 4; // 33.5 MB
  __bf16* Tt      = (__bf16*)(ws + off); off += (size_t)NB * HEADS * OUTS * INS * 2; // 16.8 MB
  __bf16* feat    = (__bf16*)(ws + off); off += (size_t)NB * OUTS * CDIM * 2;        // 16.8 MB
  (void)ws_size; (void)in_sizes; (void)n_in; (void)out_size;

  // 1) f32 -> bf16 conversions (+ transpose of x)
  convert_x_kernel<<<dim3(IND / 32, INS / 32, NB), dim3(32, 8), 0, stream>>>(x, x_bf, xT_bf);
  {
    int ne = HEADS * OUTS * IND;   // 524288
    convert_flat_kernel<<<(ne / 4 + 255) / 256, 256, 0, stream>>>(otw, otw_bf, ne);
  }
  {
    int ne = OUTD * CDIM;          // 4194304
    convert_flat_kernel<<<(ne / 4 + 255) / 256, 256, 0, stream>>>(linw, linw_bf, ne);
  }

  // 2) K = x . otw^T / eps   (16384 x 512, K=1024)
  gemm1_kernel<<<dim3((NB * INS) / 128, (HEADS * OUTS) / 128), 256, 0, stream>>>(
      x_bf, otw_bf, Kmat);

  // 3) Sinkhorn, 10 iterations; emits T^T bf16
  sinkhorn_kernel<<<NB * HEADS, 256, 0, stream>>>(Kmat, Tt);

  // 4) feat = T^T . x   (batched 128 x 1024, K=1024)
  gemm2_kernel<<<dim3(1, IND / 128, NB * HEADS), 256, 0, stream>>>(Tt, xT_bf, feat);

  // 5) out = relu(feat . linw^T + b)   (2048 x 1024, K=4096)
  gemm3_kernel<<<dim3((NB * OUTS) / 128, OUTD / 128), 256, 0, stream>>>(
      feat, linw_bf, linb, (float*)d_out);
}
